// TransformerBlock_39092792328913
// MI455X (gfx1250) — matrix operand order repaired and checked by reading, not compiled, not run on hardware
//
#include <hip/hip_runtime.h>
#include <hip/hip_bf16.h>
#include <cstdint>
#include <cstddef>

// ---------- types ----------
typedef __bf16 bf16;
typedef __attribute__((ext_vector_type(16))) __bf16 v16bf;
typedef __attribute__((ext_vector_type(8)))  float  v8f;
typedef __attribute__((ext_vector_type(4)))  int    v4i;

struct alignas(16) B128 { unsigned int w[4]; };
struct alignas(8)  B64  { unsigned int w[2]; };
struct alignas(16) F128 { float x, y, z, w; };

union FragBF { v16bf v; B128 u[2]; };
union Pack4  { bf16 h[4]; B64 u; };
union Pack16 { bf16 h[16]; B128 q[2]; };

__device__ __forceinline__ v8f wmma_bf16(const FragBF& a, const FragBF& b, v8f c) {
  // D = A(16x32 bf16) x B(32x16 bf16) + C(16x16 f32)
  return __builtin_amdgcn_wmma_f32_16x16x32_bf16(false, a.v, false, b.v, (short)0, c,
                                                 false, false);
}

__device__ __forceinline__ B128 ldg128(const bf16* p) { return *(const B128*)p; }
__device__ __forceinline__ void sts128(bf16* p, B128 v) { *(B128*)p = v; }

// ---- CDNA5 async global->LDS copy (ASYNCcnt), guarded by builtin presence ----
#if defined(__has_builtin)
#if __has_builtin(__builtin_amdgcn_global_load_async_to_lds_b128)
#define HAS_ASYNC_LDS 1
#endif
#endif
#ifndef HAS_ASYNC_LDS
#define HAS_ASYNC_LDS 0
#endif

#if HAS_ASYNC_LDS
__device__ __forceinline__ void async_copy16(const bf16* g, bf16* l) {
  __builtin_amdgcn_global_load_async_to_lds_b128(
      (__attribute__((address_space(1))) v4i*)g,
      (__attribute__((address_space(3))) v4i*)l, 0, 0);
}
__device__ __forceinline__ void async_wait0() {
#if __has_builtin(__builtin_amdgcn_s_wait_asynccnt)
  __builtin_amdgcn_s_wait_asynccnt(0);
#else
  asm volatile("s_wait_asynccnt 0x0" ::: "memory");
#endif
}
#endif

#define D_MODEL 1024
#define SEQ     2048
#define NROWS   8192   // B*S

// ---------- fp32 -> bf16 convert ----------
__global__ __launch_bounds__(256)
void cvt_bf16_k(const float* __restrict__ in, bf16* __restrict__ out, int n4) {
  int i = blockIdx.x * 256 + threadIdx.x;
  if (i < n4) {
    F128 v = *(const F128*)&in[(size_t)i * 4];
    Pack4 o;
    o.h[0] = (bf16)v.x; o.h[1] = (bf16)v.y; o.h[2] = (bf16)v.z; o.h[3] = (bf16)v.w;
    *(B64*)&out[(size_t)i * 4] = o.u;
  }
}

// ---------- RMSNorm (fp32 in, bf16 out), one row per block ----------
__global__ __launch_bounds__(256)
void rmsnorm_k(const float* __restrict__ x, const float* __restrict__ g,
               bf16* __restrict__ out) {
  const int row = blockIdx.x, tid = threadIdx.x;
  const float* xr = x + (size_t)row * D_MODEL;
  F128 v = *(const F128*)&xr[tid * 4];
  float ss = v.x * v.x + v.y * v.y + v.z * v.z + v.w * v.w;
  ss += __shfl_xor(ss, 1, 32);
  ss += __shfl_xor(ss, 2, 32);
  ss += __shfl_xor(ss, 4, 32);
  ss += __shfl_xor(ss, 8, 32);
  ss += __shfl_xor(ss, 16, 32);
  __shared__ float red[8];
  if ((tid & 31) == 0) red[tid >> 5] = ss;
  __syncthreads();
  float tot = 0.f;
#pragma unroll
  for (int i = 0; i < 8; ++i) tot += red[i];
  float rinv = rsqrtf(tot * (1.0f / (float)D_MODEL) + 1e-5f);
  F128 gv = *(const F128*)&g[tid * 4];
  Pack4 o;
  o.h[0] = (bf16)(v.x * gv.x * rinv);
  o.h[1] = (bf16)(v.y * gv.y * rinv);
  o.h[2] = (bf16)(v.z * gv.z * rinv);
  o.h[3] = (bf16)(v.w * gv.w * rinv);
  *(B64*)&out[(size_t)row * D_MODEL + tid * 4] = o.u;
}

// ---------- bf16 GEMM: C[M,N] = A[M,K] * Bw[N,K]^T (+res, gelu) ----------
// tile 128x128, BK=32, 256 threads = 8 waves, wave = 64x32 region (4x2 16x16 tiles)
// double-buffered LDS, software-pipelined staging (async-to-LDS when available)
__global__ __launch_bounds__(256)
void gemm_bf16_k(const bf16* __restrict__ A, const bf16* __restrict__ Bw,
                 const float* __restrict__ Cres, float* __restrict__ Of,
                 bf16* __restrict__ Ob, int M, int N, int K, int act_gelu) {
  constexpr int SA = 40;  // padded lds row stride (bf16 elems), 80B
  __shared__ __align__(16) bf16 As[2][128 * SA];
  __shared__ __align__(16) bf16 Bs[2][128 * SA];
  const int tid  = threadIdx.x;
  const int lane = tid & 31, wid = tid >> 5;
  const int half = lane >> 4, l16 = lane & 15;
  const int nbase = blockIdx.x * 128;
  const int mbase = blockIdx.y * 128;
  const int wm = (wid & 1) * 64;
  const int wn = (wid >> 1) * 32;

  // per-thread staging slots: 2 x 16B for A tile, 2 x 16B for B tile
  const int u0 = tid * 2, u1 = tid * 2 + 1;
  const int r0 = u0 >> 2, s0 = (u0 & 3) * 8;
  const int r1 = u1 >> 2, s1 = (u1 & 3) * 8;
  const bf16* gA0 = A  + (size_t)(mbase + r0) * K + s0;
  const bf16* gA1 = A  + (size_t)(mbase + r1) * K + s1;
  const bf16* gB0 = Bw + (size_t)(nbase + r0) * K + s0;
  const bf16* gB1 = Bw + (size_t)(nbase + r1) * K + s1;
  const int l0 = r0 * SA + s0, l1 = r1 * SA + s1;

  const v8f zero8 = {0.f, 0.f, 0.f, 0.f, 0.f, 0.f, 0.f, 0.f};
  v8f acc[4][2];
#pragma unroll
  for (int mi = 0; mi < 4; ++mi)
#pragma unroll
    for (int ni = 0; ni < 2; ++ni) acc[mi][ni] = zero8;

  auto compute = [&](int cur) {
    FragBF af[4], bfr[2];
#pragma unroll
    for (int mi = 0; mi < 4; ++mi) {
      int r = wm + mi * 16 + l16;
      af[mi].u[0] = *(const B128*)&As[cur][r * SA + 8 * half];
      af[mi].u[1] = *(const B128*)&As[cur][r * SA + 16 + 8 * half];
    }
#pragma unroll
    for (int ni = 0; ni < 2; ++ni) {
      int c = wn + ni * 16 + l16;
      bfr[ni].u[0] = *(const B128*)&Bs[cur][c * SA + 8 * half];
      bfr[ni].u[1] = *(const B128*)&Bs[cur][c * SA + 16 + 8 * half];
    }
#pragma unroll
    for (int mi = 0; mi < 4; ++mi)
#pragma unroll
      for (int ni = 0; ni < 2; ++ni)
        acc[mi][ni] = wmma_bf16(af[mi], bfr[ni], acc[mi][ni]);
  };

  const int NT = K >> 5;  // number of 32-wide K slabs

#if HAS_ASYNC_LDS
  // ---- async DMA pipeline: tiles go straight to LDS, tracked by ASYNCcnt ----
  {
    async_copy16(gA0, &As[0][l0]);
    async_copy16(gA1, &As[0][l1]);
    async_copy16(gB0, &Bs[0][l0]);
    async_copy16(gB1, &Bs[0][l1]);
    async_wait0();
  }
  __syncthreads();
  for (int t = 0; t < NT; ++t) {
    const int cur = t & 1;
    const bool more = (t + 1 < NT);
    if (more) {
      const int ko = (t + 1) * 32;
      async_copy16(gA0 + ko, &As[cur ^ 1][l0]);
      async_copy16(gA1 + ko, &As[cur ^ 1][l1]);
      async_copy16(gB0 + ko, &Bs[cur ^ 1][l0]);
      async_copy16(gB1 + ko, &Bs[cur ^ 1][l1]);
    }
    compute(cur);              // overlaps with the async copies above
    if (more) async_wait0();
    __syncthreads();
  }
#else
  // ---- register-staged pipeline: issue next-tile loads before compute ----
  {
    B128 a0 = ldg128(gA0), a1 = ldg128(gA1), b0 = ldg128(gB0), b1 = ldg128(gB1);
    sts128(&As[0][l0], a0); sts128(&As[0][l1], a1);
    sts128(&Bs[0][l0], b0); sts128(&Bs[0][l1], b1);
  }
  __syncthreads();
  for (int t = 0; t < NT; ++t) {
    const int cur = t & 1;
    const bool more = (t + 1 < NT);
    B128 na0, na1, nb0, nb1;
    if (more) {
      const int ko = (t + 1) * 32;
      na0 = ldg128(gA0 + ko); na1 = ldg128(gA1 + ko);
      nb0 = ldg128(gB0 + ko); nb1 = ldg128(gB1 + ko);
    }
    compute(cur);              // global loads above stay in flight during compute
    if (more) {
      sts128(&As[cur ^ 1][l0], na0); sts128(&As[cur ^ 1][l1], na1);
      sts128(&Bs[cur ^ 1][l0], nb0); sts128(&Bs[cur ^ 1][l1], nb1);
    }
    __syncthreads();
  }
#endif

  // epilogue: C-layout — lane holds col n = l16, rows r + 8*half
#pragma unroll
  for (int mi = 0; mi < 4; ++mi) {
#pragma unroll
    for (int ni = 0; ni < 2; ++ni) {
#pragma unroll
      for (int r = 0; r < 8; ++r) {
        int m = mbase + wm + mi * 16 + half * 8 + r;
        int n = nbase + wn + ni * 16 + l16;
        size_t idx = (size_t)m * N + n;
        float v = acc[mi][ni][r];
        if (Cres) v += Cres[idx];
        if (act_gelu) v = 0.5f * v * (1.f + erff(v * 0.70710678118654752f));
        if (Of) Of[idx] = v;
        if (Ob) Ob[idx] = (bf16)v;
      }
    }
  }
}

// ---------- flash attention (causal), qkv bf16 [8192,3072], O bf16 [8192,1024] ----------
// grid (B*H, S/128); 256 threads = 8 waves; wave owns 16 q rows; K-tile = 64
// K/V tile for iteration t+1 prefetched into registers during compute of t
__global__ __launch_bounds__(256)
void attn_k(const bf16* __restrict__ qkv, bf16* __restrict__ O) {
  constexpr int SK = 72;  // padded lds row stride (bf16), 144B
  __shared__ __align__(16) bf16 Ks[64 * SK];       // [k(64)][d(64)]
  __shared__ __align__(16) bf16 Vt[64 * SK];       // [d(64)][k(64)]  (transposed)
  __shared__ __align__(16) bf16 Ps[8 * 16 * SK];   // per-wave P strip [16][64]
  const int tid  = threadIdx.x;
  const int lane = tid & 31, w = tid >> 5;
  const int half = lane >> 4, l16 = lane & 15;
  const int bh = blockIdx.x;
  const int b = bh >> 4, h = bh & 15;
  const int qt = blockIdx.y;
  const int qbase = qt * 128;

  // Q strip (16x64) as two A-fragments, loaded once
  FragBF qa[2];
  {
    const bf16* Qr = qkv + ((size_t)(b * SEQ + qbase + w * 16 + l16)) * 3072 + h * 64;
    qa[0].u[0] = *(const B128*)&Qr[8 * half];
    qa[0].u[1] = *(const B128*)&Qr[16 + 8 * half];
    qa[1].u[0] = *(const B128*)&Qr[32 + 8 * half];
    qa[1].u[1] = *(const B128*)&Qr[48 + 8 * half];
  }

  // per-thread K/V staging slots
  const int uK0 = tid * 2, uK1 = tid * 2 + 1;
  const int rK0 = uK0 >> 3, sK0 = (uK0 & 7) * 8;
  const int rK1 = uK1 >> 3, sK1 = (uK1 & 7) * 8;
  const int vk = tid & 63, vdb = (tid >> 6) * 16;

  auto loadKV = [&](int ktb, B128& k0, B128& k1, B128& v0, B128& v1) {
    k0 = ldg128(&qkv[((size_t)(b * SEQ + ktb + rK0)) * 3072 + 1024 + h * 64 + sK0]);
    k1 = ldg128(&qkv[((size_t)(b * SEQ + ktb + rK1)) * 3072 + 1024 + h * 64 + sK1]);
    const bf16* Vr = &qkv[((size_t)(b * SEQ + ktb + vk)) * 3072 + 2048 + h * 64 + vdb];
    v0 = ldg128(Vr);
    v1 = ldg128(Vr + 8);
  };
  auto storeKV = [&](B128 k0, B128 k1, B128 v0, B128 v1) {
    sts128(&Ks[rK0 * SK + sK0], k0);
    sts128(&Ks[rK1 * SK + sK1], k1);
    Pack16 tmp;
    tmp.q[0] = v0; tmp.q[1] = v1;
#pragma unroll
    for (int j = 0; j < 16; ++j) Vt[(vdb + j) * SK + vk] = tmp.h[j];
  };

  const v8f zero8 = {0.f, 0.f, 0.f, 0.f, 0.f, 0.f, 0.f, 0.f};
  float mrun[8], lrun[8];
  v8f o[4];
#pragma unroll
  for (int r = 0; r < 8; ++r) { mrun[r] = -1e30f; lrun[r] = 0.f; }
#pragma unroll
  for (int dt = 0; dt < 4; ++dt) o[dt] = zero8;

  const int nkt = 2 * qt + 2;  // 64-wide k tiles covering [0, qbase+128)
  B128 ck0, ck1, cv0, cv1;
  loadKV(0, ck0, ck1, cv0, cv1);

  for (int kt = 0; kt < nkt; ++kt) {
    const int ktb = kt * 64;
    storeKV(ck0, ck1, cv0, cv1);   // consumes prev iteration's in-flight loads
    __syncthreads();
    if (kt + 1 < nkt) loadKV((kt + 1) * 64, ck0, ck1, cv0, cv1);  // prefetch

    // S = Q * K^T : 4 tiles of 16x16, Kdim(d)=64 -> 2 WMMA each
    v8f s[4];
#pragma unroll
    for (int nt = 0; nt < 4; ++nt) {
      FragBF kb0, kb1;
      int kc = nt * 16 + l16;
      kb0.u[0] = *(const B128*)&Ks[kc * SK + 8 * half];
      kb0.u[1] = *(const B128*)&Ks[kc * SK + 16 + 8 * half];
      kb1.u[0] = *(const B128*)&Ks[kc * SK + 32 + 8 * half];
      kb1.u[1] = *(const B128*)&Ks[kc * SK + 48 + 8 * half];
      v8f a = zero8;
      a = wmma_bf16(qa[0], kb0, a);
      a = wmma_bf16(qa[1], kb1, a);
      s[nt] = a;
    }

    // online softmax per C-layout row (r + 8*half), reduce across 16 lanes
#pragma unroll
    for (int r = 0; r < 8; ++r) {
      int qg = qbase + w * 16 + half * 8 + r;
      float rowmax = -1e30f;
#pragma unroll
      for (int nt = 0; nt < 4; ++nt) {
        int kg = ktb + nt * 16 + l16;
        float v = (kg <= qg) ? s[nt][r] * 0.125f : -1e30f;  // 1/sqrt(64)
        s[nt][r] = v;
        rowmax = fmaxf(rowmax, v);
      }
      rowmax = fmaxf(rowmax, __shfl_xor(rowmax, 1, 32));
      rowmax = fmaxf(rowmax, __shfl_xor(rowmax, 2, 32));
      rowmax = fmaxf(rowmax, __shfl_xor(rowmax, 4, 32));
      rowmax = fmaxf(rowmax, __shfl_xor(rowmax, 8, 32));
      float mnew  = fmaxf(mrun[r], rowmax);
      float alpha = __expf(mrun[r] - mnew);
      float rowsum = 0.f;
#pragma unroll
      for (int nt = 0; nt < 4; ++nt) {
        float p = __expf(s[nt][r] - mnew);
        s[nt][r] = p;
        rowsum += p;
      }
      rowsum += __shfl_xor(rowsum, 1, 32);
      rowsum += __shfl_xor(rowsum, 2, 32);
      rowsum += __shfl_xor(rowsum, 4, 32);
      rowsum += __shfl_xor(rowsum, 8, 32);
      lrun[r] = lrun[r] * alpha + rowsum;
      mrun[r] = mnew;
#pragma unroll
      for (int dt = 0; dt < 4; ++dt) o[dt][r] *= alpha;
      // store P row to wave-private LDS strip (C-layout -> memory)
#pragma unroll
      for (int nt = 0; nt < 4; ++nt)
        Ps[(w * 16 + half * 8 + r) * SK + nt * 16 + l16] = (bf16)s[nt][r];
    }

    // O += P(16x64) * V(64x64): A-frags from Ps, B-frags from Vt
#pragma unroll
    for (int kk2 = 0; kk2 < 2; ++kk2) {
      FragBF pa;
      int pr = w * 16 + l16;
      pa.u[0] = *(const B128*)&Ps[pr * SK + kk2 * 32 + 8 * half];
      pa.u[1] = *(const B128*)&Ps[pr * SK + kk2 * 32 + 16 + 8 * half];
#pragma unroll
      for (int dt = 0; dt < 4; ++dt) {
        FragBF vb;
        int d = dt * 16 + l16;
        vb.u[0] = *(const B128*)&Vt[d * SK + kk2 * 32 + 8 * half];
        vb.u[1] = *(const B128*)&Vt[d * SK + kk2 * 32 + 16 + 8 * half];
        o[dt] = wmma_bf16(pa, vb, o[dt]);
      }
    }
    __syncthreads();
  }

  // normalize and write O [row, h*64+d] bf16
#pragma unroll
  for (int dt = 0; dt < 4; ++dt) {
#pragma unroll
    for (int r = 0; r < 8; ++r) {
      int qg = qbase + w * 16 + half * 8 + r;
      float val = o[dt][r] / lrun[r];
      O[((size_t)(b * SEQ + qg)) * D_MODEL + h * 64 + dt * 16 + l16] = (bf16)val;
    }
  }
}

// ---------- host launcher ----------
extern "C" void kernel_launch(void* const* d_in, const int* in_sizes, int n_in,
                              void* d_out, int out_size, void* d_ws, size_t ws_size,
                              hipStream_t stream) {
  const float* x    = (const float*)d_in[0];
  const float* Wqkv = (const float*)d_in[1];
  const float* Wo   = (const float*)d_in[2];
  const float* w1   = (const float*)d_in[3];
  const float* w2   = (const float*)d_in[4];
  const float* g1   = (const float*)d_in[5];
  const float* g2   = (const float*)d_in[6];
  float* out = (float*)d_out;

  char* ws = (char*)d_ws;
  size_t off = 0;
  auto alloc = [&](size_t bytes) -> char* {
    char* p = ws + off;
    off += (bytes + 255) & ~(size_t)255;
    return p;
  };
  bf16*  wq_b = (bf16*)alloc(3ull * 1024 * 1024 * 2);
  bf16*  wo_b = (bf16*)alloc(1024ull * 1024 * 2);
  bf16*  w1_b = (bf16*)alloc(4096ull * 1024 * 2);
  bf16*  w2_b = (bf16*)alloc(4096ull * 1024 * 2);
  bf16*  hbuf = (bf16*)alloc((size_t)NROWS * 1024 * 2);   // rmsnorm output (reused)
  bf16*  big  = (bf16*)alloc((size_t)NROWS * 4096 * 2);   // qkv (Nx3072) then ffn hidden (Nx4096)
  bf16*  obuf = (bf16*)alloc((size_t)NROWS * 1024 * 2);   // attention output
  float* out1 = (float*)alloc((size_t)NROWS * 1024 * 4);  // x + attn (fp32 residual)

  // convert weights to bf16 (every call: deterministic, weights live in L2)
  cvt_bf16_k<<<dim3((3 * 1024 * 1024 / 4 + 255) / 256), 256, 0, stream>>>(Wqkv, wq_b, 3 * 1024 * 1024 / 4);
  cvt_bf16_k<<<dim3((1024 * 1024 / 4 + 255) / 256), 256, 0, stream>>>(Wo, wo_b, 1024 * 1024 / 4);
  cvt_bf16_k<<<dim3((4096 * 1024 / 4 + 255) / 256), 256, 0, stream>>>(w1, w1_b, 4096 * 1024 / 4);
  cvt_bf16_k<<<dim3((4096 * 1024 / 4 + 255) / 256), 256, 0, stream>>>(w2, w2_b, 4096 * 1024 / 4);

  // h1 = rmsnorm(x, g1)  (bf16)
  rmsnorm_k<<<dim3(NROWS), 256, 0, stream>>>(x, g1, hbuf);
  // qkv = h1 @ Wqkv^T   [8192,3072] bf16
  gemm_bf16_k<<<dim3(3072 / 128, NROWS / 128), 256, 0, stream>>>(
      hbuf, wq_b, nullptr, nullptr, big, NROWS, 3072, 1024, 0);
  // attention -> obuf [8192,1024] bf16
  attn_k<<<dim3(64, SEQ / 128), 256, 0, stream>>>(big, obuf);
  // out1 = x + obuf @ Wo^T   (fp32)
  gemm_bf16_k<<<dim3(1024 / 128, NROWS / 128), 256, 0, stream>>>(
      obuf, wo_b, x, out1, nullptr, NROWS, 1024, 1024, 0);
  // h2 = rmsnorm(out1, g2)  (bf16)
  rmsnorm_k<<<dim3(NROWS), 256, 0, stream>>>(out1, g2, hbuf);
  // ffh = gelu(h2 @ w1^T)   [8192,4096] bf16
  gemm_bf16_k<<<dim3(4096 / 128, NROWS / 128), 256, 0, stream>>>(
      hbuf, w1_b, nullptr, nullptr, big, NROWS, 4096, 1024, 1);
  // out = out1 + ffh @ w2^T  (fp32)
  gemm_bf16_k<<<dim3(1024 / 128, NROWS / 128), 256, 0, stream>>>(
      big, w2_b, out1, out, nullptr, NROWS, 1024, 4096, 0);
}
